// YoonBRE2TA_6390911336863
// MI455X (gfx1250) — matrix operand order repaired and checked by reading, not compiled, not run on hardware
//
#include <hip/hip_runtime.h>
#include <hip/hip_bf16.h>
#include <math.h>

// ---------------- model dims ----------------
#define BSZ   32
#define TA    512      // audio timesteps used (513 - 1)
#define TT    128
#define MFCC  80
#define MFCCP 96       // padded K
#define PROS  64
#define Hdim  512
#define G3H   1536     // 3*H
#define EMB   300
#define EMBP  320
#define NCLS  7
#define DOUT  1088     // 2H + PROS
#define AUDIO_ROW 80
#define AUDIO_T_SRC 513

typedef __bf16 bf16_t;
typedef __attribute__((ext_vector_type(16))) __bf16 v16bf;
typedef __attribute__((ext_vector_type(8)))  __bf16 v8bf;
typedef __attribute__((ext_vector_type(8)))  float  v8f;

// ---------------- WMMA fragment loaders (contiguous k = 16*g + j order, same for A and B) ----------------
// A (16x32 bf16, row-major source [.][lda]):
//   lane = g*16+m ; element j = A[m][k0 + 16*g + j]
//   => elements 0..7  = A[m][k0 + g*16     .. +7]
//      elements 8..15 = A[m][k0 + g*16 + 8 .. +7]
__device__ __forceinline__ v16bf load_a_frag(const bf16_t* A, int lda, int m0, int k0) {
  int lane = threadIdx.x & 31;
  int g = lane >> 4, m = lane & 15;
  const bf16_t* p = A + (size_t)(m0 + m) * lda + k0 + g * 16;
  v8bf lo = *(const v8bf*)p;
  v8bf hi = *(const v8bf*)(p + 8);
  v16bf f;
#pragma unroll
  for (int i = 0; i < 8; ++i) { f[i] = lo[i]; f[i + 8] = hi[i]; }
  return f;
}

// B (32x16 bf16). B[k][n] with B = W^T, W row-major [N][ldw] => B[k][n] = W[n][k].
//   lane = g*16+n ; element j = B[16*g + j][n] = W[n0+n][k0 + 16*g + j]
//   => one contiguous 32-byte load per lane.
__device__ __forceinline__ v16bf load_b_frag(const bf16_t* W, int ldw, int n0, int k0) {
  int lane = threadIdx.x & 31;
  int g = lane >> 4, n = lane & 15;
  const bf16_t* p = W + (size_t)(n0 + n) * ldw + k0 + g * 16;
  return *(const v16bf*)p;
}

// ---------------- converts ----------------
// dst[r][c] (r < rowsTotal, c < colsp) = c<cols ? src[srcRow(r)*cols + c] : 0
// srcRow = (r / Tdst) * Tsrc + (r % Tdst)   (handles the audio [:, :-1, :] slice)
__global__ void k_cvt(const float* __restrict__ src, bf16_t* __restrict__ dst,
                      int rowsTotal, int Tdst, int Tsrc, int cols, int colsp) {
  int i = blockIdx.x * blockDim.x + threadIdx.x;
  int total = rowsTotal * colsp;
  if (i >= total) return;
  int r = i / colsp, c = i - r * colsp;
  int sr = (r / Tdst) * Tsrc + (r % Tdst);
  dst[i] = (c < cols) ? (bf16_t)src[(size_t)sr * cols + c] : (bf16_t)0.0f;
}

__global__ void k_embed(const int* __restrict__ idx, const float* __restrict__ emb,
                        bf16_t* __restrict__ dst) {
  int i = blockIdx.x * blockDim.x + threadIdx.x;
  int total = BSZ * TT * EMBP;
  if (i >= total) return;
  int r = i / EMBP, c = i - r * EMBP;
  dst[i] = (c < EMB) ? (bf16_t)emb[(size_t)idx[r] * EMB + c] : (bf16_t)0.0f;
}

// ---------------- input-projection GEMM:  C(bf16) = A(bf16) @ W^T + bias ----------------
// N = 1536 fixed. grid.x = 1536/64 = 24, grid.y = M/128. block = 256 (8 waves).
// Each wave: one 16x64 strip (4 accumulators, A-fragment reused 4x).
__global__ void k_gemm_bias(const bf16_t* __restrict__ A, int lda,
                            const bf16_t* __restrict__ W, int ldw,
                            const float* __restrict__ bias,
                            bf16_t* __restrict__ C, int Kp) {
  int wave = threadIdx.x >> 5;
  int lane = threadIdx.x & 31;
  int m0 = blockIdx.y * 128 + wave * 16;
  int n0 = blockIdx.x * 64;
  v8f acc0 = {}, acc1 = {}, acc2 = {}, acc3 = {};
  for (int k0 = 0; k0 < Kp; k0 += 32) {
    v16bf af = load_a_frag(A, lda, m0, k0);
    v16bf b0 = load_b_frag(W, ldw, n0,      k0);
    v16bf b1 = load_b_frag(W, ldw, n0 + 16, k0);
    v16bf b2 = load_b_frag(W, ldw, n0 + 32, k0);
    v16bf b3 = load_b_frag(W, ldw, n0 + 48, k0);
    acc0 = __builtin_amdgcn_wmma_f32_16x16x32_bf16(false, af, false, b0, (short)0, acc0, false, false);
    acc1 = __builtin_amdgcn_wmma_f32_16x16x32_bf16(false, af, false, b1, (short)0, acc1, false, false);
    acc2 = __builtin_amdgcn_wmma_f32_16x16x32_bf16(false, af, false, b2, (short)0, acc2, false, false);
    acc3 = __builtin_amdgcn_wmma_f32_16x16x32_bf16(false, af, false, b3, (short)0, acc3, false, false);
  }
  int g = lane >> 4, n = lane & 15;
  v8f accs[4] = {acc0, acc1, acc2, acc3};
#pragma unroll
  for (int j = 0; j < 4; ++j) {
    int col = n0 + j * 16 + n;
    float bv = bias[col];
#pragma unroll
    for (int v = 0; v < 8; ++v) {
      int row = m0 + v + 8 * g;           // C/D layout: VGPR v -> M = v + 8*laneGroup
      C[(size_t)row * G3H + col] = (bf16_t)(accs[j][v] + bv);
    }
  }
}

// ---------------- recurrent GRU scan (one block per direction) ----------------
// block = 1024 threads = 32 waves. LDS: h (32x512 bf16) + Gh (32x1536 f32) = 224 KB.
// Per step: Gh = h @ Whh^T via WMMA (192 tiles / 32 waves = 6 tiles each),
// then f32 gate math, h updated in LDS, output written to original time index.
__global__ void k_gru_rec(const bf16_t* __restrict__ Gi,   // [2][32*T][1536] bf16 (bih folded)
                          const bf16_t* __restrict__ Whh,  // [2][1536][512] bf16
                          const float*  __restrict__ bvec, // [2][2][1536] f32
                          float* __restrict__ out_f32,     // [32*T][1024] or null
                          bf16_t* __restrict__ out_bf,     // [32*T][1024] or null
                          int T) {
  extern __shared__ char smem[];
  bf16_t* s_h  = (bf16_t*)smem;                          // 32*512 bf16
  float*  s_gh = (float*)(smem + BSZ * Hdim * sizeof(bf16_t)); // 32*1536 f32

  int dir  = blockIdx.x;
  int wave = threadIdx.x >> 5;
  int lane = threadIdx.x & 31;
  const bf16_t* Whh_d = Whh + (size_t)dir * G3H * Hdim;
  const bf16_t* Gi_d  = Gi  + (size_t)dir * BSZ * T * G3H;
  const float*  bhh   = bvec + (size_t)dir * 2 * G3H + G3H;

  for (int e = threadIdx.x; e < BSZ * Hdim; e += blockDim.x) s_h[e] = (bf16_t)0.0f;
  __syncthreads();

  for (int t = 0; t < T; ++t) {
    int t_idx = dir ? (T - 1 - t) : t;   // source/dest time index in original order

    // Gh = h @ Whh^T  (M=32, N=1536, K=512)
#pragma unroll 1
    for (int i = 0; i < 6; ++i) {
      int tile = wave * 6 + i;
      int mt = tile & 1, nt = tile >> 1;
      v8f acc = {};
#pragma unroll 1
      for (int k0 = 0; k0 < Hdim; k0 += 32) {
        v16bf af = load_a_frag(s_h, Hdim, mt * 16, k0);
        v16bf bf = load_b_frag(Whh_d, Hdim, nt * 16, k0);
        acc = __builtin_amdgcn_wmma_f32_16x16x32_bf16(false, af, false, bf, (short)0, acc, false, false);
      }
      int g = lane >> 4, n = lane & 15;
#pragma unroll
      for (int v = 0; v < 8; ++v)
        s_gh[(size_t)(mt * 16 + v + 8 * g) * G3H + nt * 16 + n] = acc[v];
    }
    __syncthreads();

    // gate update: 32*512 elements
    for (int e = threadIdx.x; e < BSZ * Hdim; e += blockDim.x) {
      int bb = e >> 9, j = e & 511;
      const bf16_t* gi = Gi_d + ((size_t)bb * T + t_idx) * G3H;
      float ir = (float)gi[j];
      float iz = (float)gi[Hdim + j];
      float in_ = (float)gi[2 * Hdim + j];
      const float* ghr = s_gh + (size_t)bb * G3H;
      float hr = ghr[j]            + bhh[j];
      float hz = ghr[Hdim + j]     + bhh[Hdim + j];
      float hn = ghr[2 * Hdim + j] + bhh[2 * Hdim + j];
      float r = 1.0f / (1.0f + expf(-(ir + hr)));
      float z = 1.0f / (1.0f + expf(-(iz + hz)));
      float nn = tanhf(in_ + r * hn);
      float hprev = (float)s_h[e];
      float hnew = (1.0f - z) * nn + z * hprev;
      size_t o = ((size_t)bb * T + t_idx) * (2 * Hdim) + (size_t)dir * Hdim + j;
      if (out_f32) out_f32[o] = hnew;
      if (out_bf)  out_bf[o]  = (bf16_t)hnew;
      s_h[e] = (bf16_t)hnew;
    }
    __syncthreads();
  }
}

// ---------------- attention pooling ----------------
// variant 0: A = s / sum(s)          (H1, O = text,  q = audio last)
// variant 1: A = (1+s) / (1+sum(s))  (H2, O = audio, q = text last)
__global__ void k_attn(const float* __restrict__ O,     // [B][T][1024]
                       const float* __restrict__ q_src, // [B][Tq][1024]
                       const float* __restrict__ i_audio,
                       int T, int Tq, int variant,
                       float* __restrict__ Hout) {      // [B][1088]
  __shared__ float q[DOUT];
  __shared__ float s[TA];
  __shared__ float red[256];
  int b = blockIdx.x, tid = threadIdx.x;
  const float* pros = i_audio + ((size_t)b * AUDIO_T_SRC + (AUDIO_T_SRC - 1)) * AUDIO_ROW;
  for (int d = tid; d < DOUT; d += 256)
    q[d] = (d < 2 * Hdim) ? q_src[((size_t)b * Tq + (Tq - 1)) * (2 * Hdim) + d]
                          : pros[d - 2 * Hdim];
  __syncthreads();
  float ppart = 0.0f;
  for (int p = 0; p < PROS; ++p) ppart += q[2 * Hdim + p] * pros[p];
  for (int t = tid; t < T; t += 256) {
    const float* o = O + ((size_t)b * T + t) * (2 * Hdim);
    float dot = ppart;
    for (int d = 0; d < 2 * Hdim; ++d) dot += q[d] * o[d];
    s[t] = dot;
  }
  __syncthreads();
  float loc = 0.0f;
  for (int t = tid; t < T; t += 256) loc += s[t];
  red[tid] = loc;
  __syncthreads();
  for (int st = 128; st > 0; st >>= 1) { if (tid < st) red[tid] += red[tid + st]; __syncthreads(); }
  float S = red[0];
  float denom = variant ? (1.0f + S) : S;
  float sumA  = variant ? ((float)T + S) / denom : 1.0f;
  for (int d = tid; d < DOUT; d += 256) {
    if (d < 2 * Hdim) {
      float acc = 0.0f;
      for (int t = 0; t < T; ++t) {
        float a = (variant ? (1.0f + s[t]) : s[t]) / denom;
        acc += a * O[((size_t)b * T + t) * (2 * Hdim) + d];
      }
      Hout[(size_t)b * DOUT + d] = acc;
    } else {
      Hout[(size_t)b * DOUT + d] = pros[d - 2 * Hdim] * sumA;
    }
  }
}

// ---------------- FC + softmax ----------------
__global__ void k_fc(const float* __restrict__ H1, const float* __restrict__ H2,
                     const float* __restrict__ fw, const float* __restrict__ fb,
                     float* __restrict__ out) {
  __shared__ float lg[NCLS];
  int b = blockIdx.x, tid = threadIdx.x;
  if (tid < NCLS) {
    const float* w = fw + (size_t)tid * (2 * DOUT);
    float a = fb[tid];
    for (int d = 0; d < DOUT; ++d) a += w[d] * H1[(size_t)b * DOUT + d];
    for (int d = 0; d < DOUT; ++d) a += w[DOUT + d] * H2[(size_t)b * DOUT + d];
    lg[tid] = a;
  }
  __syncthreads();
  if (tid < NCLS) {
    float mx = lg[0];
    for (int i = 1; i < NCLS; ++i) mx = fmaxf(mx, lg[i]);
    float sm = 0.0f;
    for (int i = 0; i < NCLS; ++i) sm += expf(lg[i] - mx);
    out[(size_t)b * NCLS + tid] = expf(lg[tid] - mx) / sm;
  }
}

// ---------------- host launch ----------------
static inline size_t alloc_off(size_t& cur, size_t bytes) {
  size_t o = cur;
  cur += (bytes + 255) & ~(size_t)255;
  return o;
}

extern "C" void kernel_launch(void* const* d_in, const int* in_sizes, int n_in,
                              void* d_out, int out_size, void* d_ws, size_t ws_size,
                              hipStream_t stream) {
  (void)in_sizes; (void)n_in; (void)out_size; (void)ws_size;
  const float* i_audio = (const float*)d_in[0];
  const int*   i_text  = (const int*)d_in[1];
  const float* emb     = (const float*)d_in[2];
  const float* fc_w    = (const float*)d_in[3];
  const float* fc_b    = (const float*)d_in[4];
  const float* a_Wih0  = (const float*)d_in[5];
  const float* a_Whh0  = (const float*)d_in[6];
  const float* a_b0    = (const float*)d_in[7];
  const float* a_Wih1  = (const float*)d_in[8];
  const float* a_Whh1  = (const float*)d_in[9];
  const float* a_b1    = (const float*)d_in[10];
  const float* t_Wih0  = (const float*)d_in[11];
  const float* t_Whh0  = (const float*)d_in[12];
  const float* t_b0    = (const float*)d_in[13];
  const float* t_Wih1  = (const float*)d_in[14];
  const float* t_Whh1  = (const float*)d_in[15];
  const float* t_b1    = (const float*)d_in[16];
  float* out = (float*)d_out;

  char* ws = (char*)d_ws;
  size_t cur = 0;
  bf16_t* aW0  = (bf16_t*)(ws + alloc_off(cur, (size_t)2 * G3H * MFCCP * 2));
  bf16_t* aWH0 = (bf16_t*)(ws + alloc_off(cur, (size_t)2 * G3H * Hdim * 2));
  bf16_t* aW1  = (bf16_t*)(ws + alloc_off(cur, (size_t)2 * G3H * 2 * Hdim * 2));
  bf16_t* aWH1 = (bf16_t*)(ws + alloc_off(cur, (size_t)2 * G3H * Hdim * 2));
  bf16_t* tW0  = (bf16_t*)(ws + alloc_off(cur, (size_t)2 * G3H * EMBP * 2));
  bf16_t* tWH0 = (bf16_t*)(ws + alloc_off(cur, (size_t)2 * G3H * Hdim * 2));
  bf16_t* tW1  = (bf16_t*)(ws + alloc_off(cur, (size_t)2 * G3H * 2 * Hdim * 2));
  bf16_t* tWH1 = (bf16_t*)(ws + alloc_off(cur, (size_t)2 * G3H * Hdim * 2));
  bf16_t* aX   = (bf16_t*)(ws + alloc_off(cur, (size_t)BSZ * TA * MFCCP * 2));
  bf16_t* tX   = (bf16_t*)(ws + alloc_off(cur, (size_t)BSZ * TT * EMBP * 2));
  bf16_t* aO0b = (bf16_t*)(ws + alloc_off(cur, (size_t)BSZ * TA * 2 * Hdim * 2));
  bf16_t* tO0b = (bf16_t*)(ws + alloc_off(cur, (size_t)BSZ * TT * 2 * Hdim * 2));
  float*  aO1f = (float*) (ws + alloc_off(cur, (size_t)BSZ * TA * 2 * Hdim * 4));
  float*  tO1f = (float*) (ws + alloc_off(cur, (size_t)BSZ * TT * 2 * Hdim * 4));
  bf16_t* GiA  = (bf16_t*)(ws + alloc_off(cur, (size_t)2 * BSZ * TA * G3H * 2));
  bf16_t* GiT  = (bf16_t*)(ws + alloc_off(cur, (size_t)2 * BSZ * TT * G3H * 2));
  float*  H1   = (float*) (ws + alloc_off(cur, (size_t)BSZ * DOUT * 4));
  float*  H2   = (float*) (ws + alloc_off(cur, (size_t)BSZ * DOUT * 4));

  const int SMEM_REC = BSZ * Hdim * 2 + BSZ * G3H * 4; // 32KB h + 196KB Gh = 229376B
  static int attr_set = 0; // idempotent attribute, not a stream op
  if (!attr_set) {
    hipFuncSetAttribute(reinterpret_cast<const void*>(&k_gru_rec),
                        hipFuncAttributeMaxDynamicSharedMemorySize, SMEM_REC);
    attr_set = 1;
  }

  auto cvt = [&](const float* s, bf16_t* d, int rows, int Tdst, int Tsrc, int cols, int colsp) {
    int total = rows * colsp;
    k_cvt<<<(total + 255) / 256, 256, 0, stream>>>(s, d, rows, Tdst, Tsrc, cols, colsp);
  };

  // weight + activation converts
  cvt(a_Wih0, aW0,  2 * G3H, 1, 1, MFCC, MFCCP);
  cvt(a_Whh0, aWH0, 2 * G3H, 1, 1, Hdim, Hdim);
  cvt(a_Wih1, aW1,  2 * G3H, 1, 1, 2 * Hdim, 2 * Hdim);
  cvt(a_Whh1, aWH1, 2 * G3H, 1, 1, Hdim, Hdim);
  cvt(t_Wih0, tW0,  2 * G3H, 1, 1, EMB, EMBP);
  cvt(t_Whh0, tWH0, 2 * G3H, 1, 1, Hdim, Hdim);
  cvt(t_Wih1, tW1,  2 * G3H, 1, 1, 2 * Hdim, 2 * Hdim);
  cvt(t_Whh1, tWH1, 2 * G3H, 1, 1, Hdim, Hdim);
  cvt(i_audio, aX, BSZ * TA, TA, AUDIO_T_SRC, MFCC, MFCCP);   // drops last frame
  {
    int total = BSZ * TT * EMBP;
    k_embed<<<(total + 255) / 256, 256, 0, stream>>>(i_text, emb, tX);
  }

  dim3 blk(256);
  // ---- audio layer 0 ----
  for (int d = 0; d < 2; ++d)
    k_gemm_bias<<<dim3(24, BSZ * TA / 128), blk, 0, stream>>>(
        aX, MFCCP, aW0 + (size_t)d * G3H * MFCCP, MFCCP,
        a_b0 + (size_t)d * 2 * G3H, GiA + (size_t)d * BSZ * TA * G3H, MFCCP);
  k_gru_rec<<<2, 1024, SMEM_REC, stream>>>(GiA, aWH0, a_b0, nullptr, aO0b, TA);
  // ---- audio layer 1 ----
  for (int d = 0; d < 2; ++d)
    k_gemm_bias<<<dim3(24, BSZ * TA / 128), blk, 0, stream>>>(
        aO0b, 2 * Hdim, aW1 + (size_t)d * G3H * 2 * Hdim, 2 * Hdim,
        a_b1 + (size_t)d * 2 * G3H, GiA + (size_t)d * BSZ * TA * G3H, 2 * Hdim);
  k_gru_rec<<<2, 1024, SMEM_REC, stream>>>(GiA, aWH1, a_b1, aO1f, nullptr, TA);
  // ---- text layer 0 ----
  for (int d = 0; d < 2; ++d)
    k_gemm_bias<<<dim3(24, BSZ * TT / 128), blk, 0, stream>>>(
        tX, EMBP, tW0 + (size_t)d * G3H * EMBP, EMBP,
        t_b0 + (size_t)d * 2 * G3H, GiT + (size_t)d * BSZ * TT * G3H, EMBP);
  k_gru_rec<<<2, 1024, SMEM_REC, stream>>>(GiT, tWH0, t_b0, nullptr, tO0b, TT);
  // ---- text layer 1 ----
  for (int d = 0; d < 2; ++d)
    k_gemm_bias<<<dim3(24, BSZ * TT / 128), blk, 0, stream>>>(
        tO0b, 2 * Hdim, tW1 + (size_t)d * G3H * 2 * Hdim, 2 * Hdim,
        t_b1 + (size_t)d * 2 * G3H, GiT + (size_t)d * BSZ * TT * G3H, 2 * Hdim);
  k_gru_rec<<<2, 1024, SMEM_REC, stream>>>(GiT, tWH1, t_b1, tO1f, nullptr, TT);

  // ---- attention + classifier ----
  k_attn<<<BSZ, 256, 0, stream>>>(tO1f, aO1f, i_audio, TT, TA, 0, H1);
  k_attn<<<BSZ, 256, 0, stream>>>(aO1f, tO1f, i_audio, TA, TT, 1, H2);
  k_fc<<<BSZ, 32, 0, stream>>>(H1, H2, fc_w, fc_b, out);
}
